// TransformerBlock_72894184947839
// MI455X (gfx1250) — matrix operand order repaired and checked by reading, not compiled, not run on hardware
//
#include <hip/hip_runtime.h>

// ---------------- problem constants ----------------
#define BB 64
#define TT 256
#define DD 1024
#define HH 16
#define HSZ 64
#define FFD 4096
#define MM (BB * TT)   // 16384 rows

typedef __attribute__((ext_vector_type(16))) __bf16 v16bf;
typedef __attribute__((ext_vector_type(8)))  float  v8f;

struct H8 { unsigned int a, b, c, d; };          // 16 bytes = 8 bf16
struct F2 { H8 lo, hi; };                         // 32 bytes = 16 bf16
struct B16 { __bf16 e[16]; };                     // 32 bytes

__device__ __forceinline__ __bf16 f2bf(float f) {
    unsigned u = __builtin_bit_cast(unsigned, f);
    unsigned r = (u + 0x7FFFu + ((u >> 16) & 1u)) >> 16;
    unsigned short s = (unsigned short)r;
    return __builtin_bit_cast(__bf16, s);
}

// ---------------- LayerNorm: fp32 in -> fp32 out + bf16 out ----------------
__global__ void __launch_bounds__(256)
ln_kernel(const float* __restrict__ in, const float* __restrict__ g,
          const float* __restrict__ bb, float* __restrict__ outf,
          __bf16* __restrict__ outb) {
    int row = blockIdx.x;
    int tid = threadIdx.x;
    const float* r = in + (size_t)row * DD;
    float4 v = ((const float4*)r)[tid];
    float s  = v.x + v.y + v.z + v.w;
    float s2 = v.x * v.x + v.y * v.y + v.z * v.z + v.w * v.w;
    __shared__ float rs[256], rs2[256];
    rs[tid] = s; rs2[tid] = s2;
    __syncthreads();
    for (int off = 128; off > 0; off >>= 1) {
        if (tid < off) { rs[tid] += rs[tid + off]; rs2[tid] += rs2[tid + off]; }
        __syncthreads();
    }
    float mean = rs[0] * (1.0f / DD);
    float var  = rs2[0] * (1.0f / DD) - mean * mean;
    float rstd = rsqrtf(var + 1e-5f);
    float vv[4] = {v.x, v.y, v.z, v.w};
    for (int j = 0; j < 4; ++j) {
        int col = tid * 4 + j;
        float o = (vv[j] - mean) * rstd * g[col] + bb[col];
        outf[(size_t)row * DD + col] = o;
        outb[(size_t)row * DD + col] = f2bf(o);
    }
}

// ---------------- weight conversion kernels ----------------
// Wq (H, D, HS) fp32 -> WT (H*HS, D) bf16  (N x K, row-major)
__global__ void __launch_bounds__(256)
convert_qkv_w(const float* __restrict__ W, __bf16* __restrict__ WT) {
    size_t i = (size_t)blockIdx.x * 256 + threadIdx.x;   // over (H*HS)*D
    int n = (int)(i / DD);
    int d = (int)(i % DD);
    int h = n / HSZ, e = n % HSZ;
    WT[i] = f2bf(W[(size_t)h * DD * HSZ + (size_t)d * HSZ + e]);
}

// W (K, N) fp32 row-major -> WT (N, K) bf16 row-major
__global__ void __launch_bounds__(256)
transpose_w(const float* __restrict__ W, __bf16* __restrict__ WT, int K, int N) {
    size_t i = (size_t)blockIdx.x * 256 + threadIdx.x;   // over N*K
    int n  = (int)(i / K);
    int kk = (int)(i % K);
    WT[i] = f2bf(W[(size_t)kk * N + n]);
}

// ---------------- generic WMMA bf16 GEMM ----------------
// C(M,N) = A(M,K) @ B(K,N), B given transposed (BT = N x K).
// Block: 256 threads = 8 waves arranged 4(M) x 2(N).
// Each wave: 32x64 output (2 M-subtiles x 4 N-subtiles = 8 WMMA accumulators),
// so per 32-deep K-step: 12 b128 fragment loads feed 8 WMMAs.
template <bool HAS_BIAS, bool RELU, bool HAS_RES, bool OUT_BF16>
__global__ void __launch_bounds__(256)
gemm_wmma(const __bf16* __restrict__ A, const __bf16* __restrict__ BT,
          const float* __restrict__ bias, const float* __restrict__ resid,
          float* __restrict__ Cf, __bf16* __restrict__ Cb,
          int M, int N, int K) {
    int lane = threadIdx.x & 31;
    int w    = threadIdx.x >> 5;
    int wm = w >> 1;          // 0..3
    int wn = w & 1;           // 0..1
    int M0 = blockIdx.y * 128 + wm * 32;
    int N0 = blockIdx.x * 128 + wn * 64;
    int m   = lane & 15;
    bool hiL = lane >= 16;
    int klo = hiL ? 8 : 0;    // unused here; A fragment also uses kb
    int kb  = hiL ? 16 : 0;   // B fragment K base

    const __bf16* arow0 = A  + (size_t)(M0 + m) * K;
    const __bf16* arow1 = arow0 + (size_t)16 * K;
    const __bf16* brow0 = BT + (size_t)(N0 + m) * K;
    const __bf16* brow1 = brow0 + (size_t)16 * K;
    const __bf16* brow2 = brow0 + (size_t)32 * K;
    const __bf16* brow3 = brow0 + (size_t)48 * K;

    v8f acc[2][4] = {};
    for (int k0 = 0; k0 < K; k0 += 32) {
        if (k0 + 128 < K) {
            __builtin_prefetch(arow0 + k0 + 128, 0, 1);
            __builtin_prefetch(arow1 + k0 + 128, 0, 1);
            __builtin_prefetch(brow0 + k0 + 128, 0, 1);
            __builtin_prefetch(brow2 + k0 + 128, 0, 1);
        }
        v16bf af[2], bf[4];
        {
            H8 x0 = *(const H8*)(arow0 + k0 + kb);
            H8 x1 = *(const H8*)(arow0 + k0 + kb + 8);
            af[0] = __builtin_bit_cast(v16bf, F2{x0, x1});
            H8 y0 = *(const H8*)(arow1 + k0 + kb);
            H8 y1 = *(const H8*)(arow1 + k0 + kb + 8);
            af[1] = __builtin_bit_cast(v16bf, F2{y0, y1});
        }
        {
            H8 b0 = *(const H8*)(brow0 + k0 + kb);
            H8 b1 = *(const H8*)(brow0 + k0 + kb + 8);
            bf[0] = __builtin_bit_cast(v16bf, F2{b0, b1});
            H8 c0 = *(const H8*)(brow1 + k0 + kb);
            H8 c1 = *(const H8*)(brow1 + k0 + kb + 8);
            bf[1] = __builtin_bit_cast(v16bf, F2{c0, c1});
            H8 d0 = *(const H8*)(brow2 + k0 + kb);
            H8 d1 = *(const H8*)(brow2 + k0 + kb + 8);
            bf[2] = __builtin_bit_cast(v16bf, F2{d0, d1});
            H8 e0 = *(const H8*)(brow3 + k0 + kb);
            H8 e1 = *(const H8*)(brow3 + k0 + kb + 8);
            bf[3] = __builtin_bit_cast(v16bf, F2{e0, e1});
        }
#pragma unroll
        for (int mt = 0; mt < 2; ++mt)
#pragma unroll
            for (int nt = 0; nt < 4; ++nt)
                acc[mt][nt] = __builtin_amdgcn_wmma_f32_16x16x32_bf16(
                    false, af[mt], false, bf[nt], (short)0, acc[mt][nt],
                    false, false);
    }

    int mb = hiL ? 8 : 0;
#pragma unroll
    for (int nt = 0; nt < 4; ++nt) {
        int col = N0 + nt * 16 + m;
        float bv = 0.0f;
        if (HAS_BIAS) bv = bias[col];
#pragma unroll
        for (int mt = 0; mt < 2; ++mt) {
#pragma unroll
            for (int r = 0; r < 8; ++r) {
                float cval = acc[mt][nt][r] + bv;
                if (RELU) cval = fmaxf(cval, 0.0f);
                size_t idx = (size_t)(M0 + mt * 16 + mb + r) * N + col;
                if (HAS_RES) cval += resid[idx];
                if (OUT_BF16) Cb[idx] = f2bf(cval);
                else          Cf[idx] = cval;
            }
        }
    }
}

// ---------------- fused causal attention ----------------
// One wave per (b, h, 16-row q strip). q/k/v are bf16 in (B*T, H*HS) layout.
__global__ void __launch_bounds__(32)
attn_kernel(const __bf16* __restrict__ q, const __bf16* __restrict__ k,
            const __bf16* __restrict__ v, __bf16* __restrict__ attn) {
    __shared__ __align__(16) float  ssc[16][264];
    __shared__ __align__(16) __bf16 pmat[16][272];
    __shared__ __align__(16) __bf16 vtile[32][64];

    int lane = threadIdx.x;
    int qt = blockIdx.x, h = blockIdx.y, b = blockIdx.z;
    int t0 = qt * 16;
    int m   = lane & 15;
    bool hiL = lane >= 16;
    int klo = hiL ? 8 : 0;
    int kb  = hiL ? 16 : 0;
    int mb  = hiL ? 8 : 0;

    const __bf16* qrow = q + ((size_t)(b * TT + t0 + m) * DD + h * HSZ);

    // ---- phase 1: scores for all causal s-tiles ----
    for (int st = 0; st <= qt; ++st) {
        int s0 = st * 16;
        const __bf16* krow = k + ((size_t)(b * TT + s0 + m) * DD + h * HSZ);
        v8f acc = {};
#pragma unroll
        for (int k0 = 0; k0 < HSZ; k0 += 32) {
            H8 a0 = *(const H8*)(qrow + k0 + kb);
            H8 a1 = *(const H8*)(qrow + k0 + kb + 8);
            H8 b0 = *(const H8*)(krow + k0 + kb);
            H8 b1 = *(const H8*)(krow + k0 + kb + 8);
            v16bf af = __builtin_bit_cast(v16bf, F2{a0, a1});
            v16bf bf = __builtin_bit_cast(v16bf, F2{b0, b1});
            acc = __builtin_amdgcn_wmma_f32_16x16x32_bf16(
                false, af, false, bf, (short)0, acc, false, false);
        }
        int scol = s0 + m;
#pragma unroll
        for (int r = 0; r < 8; ++r) {
            int t = t0 + mb + r;
            float val = acc[r] * 0.03125f;          // D^-0.5 = 1/32
            if (scol > t) val = -1e30f;             // causal mask
            ssc[mb + r][scol] = val;
        }
    }
    __syncthreads();

    // ---- phase 2: softmax rows (lanes 0..15, one row each) ----
    int ncol = t0 + 16;
    int kpad = ((ncol + 31) / 32) * 32;
    if (lane < 16) {
        float mx = -1e30f;
        for (int c = 0; c < ncol; ++c) mx = fmaxf(mx, ssc[lane][c]);
        float sum = 0.0f;
        for (int c = 0; c < ncol; ++c) {
            float e = __expf(ssc[lane][c] - mx);
            ssc[lane][c] = e;
            sum += e;
        }
        float inv = 1.0f / sum;
        for (int c = 0; c < ncol; ++c) pmat[lane][c] = f2bf(ssc[lane][c] * inv);
        for (int c = ncol; c < kpad; ++c) pmat[lane][c] = f2bf(0.0f);
    }
    __syncthreads();

    // ---- phase 3: attn = P @ V, V staged through LDS ----
    v8f accv[4] = {};
    for (int k0 = 0; k0 < kpad; k0 += 32) {
        __syncthreads();
        {
            int s = k0 + lane;
            const __bf16* vr = v + ((size_t)(b * TT + s) * DD + h * HSZ);
#pragma unroll
            for (int j = 0; j < 8; ++j)
                *(H8*)&vtile[lane][j * 8] = *(const H8*)(vr + j * 8);
        }
        __syncthreads();
        H8 a0 = *(const H8*)&pmat[m][k0 + klo];
        H8 a1 = *(const H8*)&pmat[m][k0 + 16 + klo];
        v16bf af = __builtin_bit_cast(v16bf, F2{a0, a1});
#pragma unroll
        for (int nt = 0; nt < 4; ++nt) {
            int N0 = nt * 16;
            B16 bt;
#pragma unroll
            for (int i = 0; i < 16; ++i) bt.e[i] = vtile[klo + i + (i >= 8 ? 8 : 0)][N0 + m];
            v16bf bf = __builtin_bit_cast(v16bf, bt);
            accv[nt] = __builtin_amdgcn_wmma_f32_16x16x32_bf16(
                false, af, false, bf, (short)0, accv[nt], false, false);
        }
    }
#pragma unroll
    for (int nt = 0; nt < 4; ++nt) {
        int col = h * HSZ + nt * 16 + m;
#pragma unroll
        for (int r = 0; r < 8; ++r)
            attn[(size_t)(b * TT + t0 + mb + r) * DD + col] = f2bf(accv[nt][r]);
    }
}

// ---------------- host-side launch ----------------
extern "C" void kernel_launch(void* const* d_in, const int* in_sizes, int n_in,
                              void* d_out, int out_size, void* d_ws, size_t ws_size,
                              hipStream_t stream) {
    const float* x     = (const float*)d_in[0];
    const float* ln1_g = (const float*)d_in[1];
    const float* ln1_b = (const float*)d_in[2];
    const float* Wq    = (const float*)d_in[3];
    const float* Wk    = (const float*)d_in[4];
    const float* Wv    = (const float*)d_in[5];
    const float* Wproj = (const float*)d_in[6];
    const float* bproj = (const float*)d_in[7];
    const float* ln2_g = (const float*)d_in[8];
    const float* ln2_b = (const float*)d_in[9];
    const float* W1    = (const float*)d_in[10];
    const float* b1    = (const float*)d_in[11];
    const float* W2    = (const float*)d_in[12];
    const float* b2    = (const float*)d_in[13];
    float* out = (float*)d_out;

    // workspace layout
    char* p = (char*)d_ws;
    auto alloc = [&](size_t bytes) {
        char* r = p;
        p += (bytes + 255) & ~(size_t)255;
        return r;
    };
    float*  h_f     = (float*) alloc((size_t)MM * DD * 4);   // also used as y (in-place residual)
    __bf16* h_bf    = (__bf16*)alloc((size_t)MM * DD * 2);
    __bf16* q_bf    = (__bf16*)alloc((size_t)MM * DD * 2);
    __bf16* k_bf    = (__bf16*)alloc((size_t)MM * DD * 2);
    __bf16* v_bf    = (__bf16*)alloc((size_t)MM * DD * 2);
    __bf16* attn_bf = (__bf16*)alloc((size_t)MM * DD * 2);
    float*  h2_f    = (float*) alloc((size_t)MM * DD * 4);
    __bf16* h2_bf   = (__bf16*)alloc((size_t)MM * DD * 2);
    __bf16* ff_bf   = (__bf16*)alloc((size_t)MM * FFD * 2);
    __bf16* WqT     = (__bf16*)alloc((size_t)DD * DD * 2);
    __bf16* WkT     = (__bf16*)alloc((size_t)DD * DD * 2);
    __bf16* WvT     = (__bf16*)alloc((size_t)DD * DD * 2);
    __bf16* WprojT  = (__bf16*)alloc((size_t)DD * DD * 2);
    __bf16* W1T     = (__bf16*)alloc((size_t)DD * FFD * 2);
    __bf16* W2T     = (__bf16*)alloc((size_t)FFD * DD * 2);

    // 1) weight conversion (bf16, transposed to N x K)
    int nW = DD * DD;
    convert_qkv_w<<<nW / 256, 256, 0, stream>>>(Wq, WqT);
    convert_qkv_w<<<nW / 256, 256, 0, stream>>>(Wk, WkT);
    convert_qkv_w<<<nW / 256, 256, 0, stream>>>(Wv, WvT);
    transpose_w<<<nW / 256, 256, 0, stream>>>(Wproj, WprojT, DD, DD);
    transpose_w<<<(DD * FFD) / 256, 256, 0, stream>>>(W1, W1T, DD, FFD);
    transpose_w<<<(FFD * DD) / 256, 256, 0, stream>>>(W2, W2T, FFD, DD);

    // 2) ln1
    ln_kernel<<<MM, 256, 0, stream>>>(x, ln1_g, ln1_b, h_f, h_bf);

    // 3) Q, K, V projections (M x 1024, K = 1024)
    dim3 gq(DD / 128, MM / 128);
    gemm_wmma<false, false, false, true><<<gq, 256, 0, stream>>>(
        h_bf, WqT, nullptr, nullptr, nullptr, q_bf, MM, DD, DD);
    gemm_wmma<false, false, false, true><<<gq, 256, 0, stream>>>(
        h_bf, WkT, nullptr, nullptr, nullptr, k_bf, MM, DD, DD);
    gemm_wmma<false, false, false, true><<<gq, 256, 0, stream>>>(
        h_bf, WvT, nullptr, nullptr, nullptr, v_bf, MM, DD, DD);

    // 4) fused causal attention
    dim3 ga(TT / 16, HH, BB);
    attn_kernel<<<ga, 32, 0, stream>>>(q_bf, k_bf, v_bf, attn_bf);

    // 5) output projection + bias + residual (y = h + attn@Wproj + bproj), in place into h_f
    gemm_wmma<true, false, true, false><<<gq, 256, 0, stream>>>(
        attn_bf, WprojT, bproj, h_f, h_f, nullptr, MM, DD, DD);

    // 6) ln2
    ln_kernel<<<MM, 256, 0, stream>>>(h_f, ln2_g, ln2_b, h2_f, h2_bf);

    // 7) FF1 with ReLU (M x 4096)
    dim3 gf1(FFD / 128, MM / 128);
    gemm_wmma<true, true, false, true><<<gf1, 256, 0, stream>>>(
        h2_bf, W1T, b1, nullptr, nullptr, ff_bf, MM, FFD, DD);

    // 8) FF2 + bias + residual -> out
    gemm_wmma<true, false, true, false><<<gq, 256, 0, stream>>>(
        ff_bf, W2T, b2, h2_f, out, nullptr, MM, DD, FFD);
}
